// TwoLevelSoftmaxMappingModule_63067299774938
// MI455X (gfx1250) — matrix operand order repaired and checked by reading, not compiled, not run on hardware
//
#include <hip/hip_runtime.h>
#include <hip/hip_bf16.h>
#include <hip/hip_fp16.h>

typedef __attribute__((ext_vector_type(16))) _Float16 v16h;
typedef __attribute__((ext_vector_type(8)))  _Float16 h8;
typedef __attribute__((ext_vector_type(16))) __bf16   v16bf;
typedef __attribute__((ext_vector_type(8)))  float    v8f;
typedef __attribute__((ext_vector_type(4)))  float    f32x4;

#define NEG_BIG (-3.0e38f)

// -----------------------------------------------------------------------------
// One-time f32 -> f16 conversion (weights are reused 256x by the GEMM, so a
// single HBM pass here removes all inner-loop cvts and halves L2 traffic).
// -----------------------------------------------------------------------------
__global__ __launch_bounds__(256)
void cvt_f32_f16(const float* __restrict__ src, _Float16* __restrict__ dst,
                 long long n)
{
    long long i = ((long long)blockIdx.x * blockDim.x + threadIdx.x) * 8;
    if (i + 8 <= n) {
        f32x4 a = *reinterpret_cast<const f32x4*>(src + i);
        f32x4 b = *reinterpret_cast<const f32x4*>(src + i + 4);
        h8 o;
        o[0] = (_Float16)a[0]; o[1] = (_Float16)a[1];
        o[2] = (_Float16)a[2]; o[3] = (_Float16)a[3];
        o[4] = (_Float16)b[0]; o[5] = (_Float16)b[1];
        o[6] = (_Float16)b[2]; o[7] = (_Float16)b[3];
        *reinterpret_cast<h8*>(dst + i) = o;
    } else {
        for (long long e = i; e < n; ++e) dst[e] = (_Float16)src[e];
    }
}

// -----------------------------------------------------------------------------
// Main path: pre-converted f16 operands. Each wave owns a 32x64 logits tile:
// 2 M-subtiles x 4 N-subtiles, 8 v_wmma_f32_16x16x32_f16 per k-step fed by
// 8 b128 loads (B fragments shared across the two M-subtiles). Fused per-row
// (max, sum(exp)) partial reduction. partial layout: [NC][B][2].
// -----------------------------------------------------------------------------
__global__ __launch_bounds__(256)
void gemm_lse_partial_h(const _Float16* __restrict__ Xh,  // [B][K]
                        const _Float16* __restrict__ Wh,  // [N][K]
                        const float* __restrict__ bias,   // [N]
                        int N, int K, int B,
                        float* __restrict__ partial)
{
    const int mTile = blockIdx.x;         // 32 rows of the batch
    const int chunk = blockIdx.y;         // 512 columns of N
    const int wave  = threadIdx.x >> 5;   // 0..7
    const int lane  = threadIdx.x & 31;
    const int half  = lane >> 4;          // 0 | 1
    const int l15   = lane & 15;

    const int rowBase = mTile * 32;
    const int colBase = chunk * 512 + wave * 64;

    v8f acc[2][4];
    #pragma unroll
    for (int mt = 0; mt < 2; ++mt)
        #pragma unroll
        for (int j = 0; j < 4; ++j)
            #pragma unroll
            for (int r = 0; r < 8; ++r) acc[mt][j][r] = 0.0f;

    // A fragment (this code's contiguous k order, not the ISA table): lane holds M=(l&15);
    // lanes 0-15 hold K=0..15, lanes 16-31 K=16..31 (matches B).
    const _Float16* xrow0 = Xh + (size_t)(rowBase + l15) * K;
    const _Float16* xrow1 = xrow0 + (size_t)16 * K;

    // B fragment: lane holds N-col=(l&15); lanes 0-15 K=0..15, 16-31 K=16..31.
    const _Float16* wrow[4];
    #pragma unroll
    for (int j = 0; j < 4; ++j) {
        int n = colBase + j * 16 + l15;
        if (n >= N) n = N - 1;            // clamp; masked out after GEMM
        wrow[j] = Wh + (size_t)n * K;
    }

    for (int ks = 0; ks < K / 32; ++ks) {
        const int kA = ks * 32 + 16 * half;
        h8 a0l = *reinterpret_cast<const h8*>(xrow0 + kA);
        h8 a0h = *reinterpret_cast<const h8*>(xrow0 + kA + 8);
        h8 a1l = *reinterpret_cast<const h8*>(xrow1 + kA);
        h8 a1h = *reinterpret_cast<const h8*>(xrow1 + kA + 8);
        v16h a0 = __builtin_shufflevector(a0l, a0h, 0,1,2,3,4,5,6,7,8,9,10,11,12,13,14,15);
        v16h a1 = __builtin_shufflevector(a1l, a1h, 0,1,2,3,4,5,6,7,8,9,10,11,12,13,14,15);

        const int kB = ks * 32 + 16 * half;
        #pragma unroll
        for (int j = 0; j < 4; ++j) {
            __builtin_prefetch(wrow[j] + kB + 128, 0, 1);   // global_prefetch_b8
            h8 bl = *reinterpret_cast<const h8*>(wrow[j] + kB);
            h8 bh = *reinterpret_cast<const h8*>(wrow[j] + kB + 8);
            v16h bf = __builtin_shufflevector(bl, bh, 0,1,2,3,4,5,6,7,8,9,10,11,12,13,14,15);
            acc[0][j] = __builtin_amdgcn_wmma_f32_16x16x32_f16(
                false, a0, false, bf, (short)0, acc[0][j], false, false);
            acc[1][j] = __builtin_amdgcn_wmma_f32_16x16x32_f16(
                false, a1, false, bf, (short)0, acc[1][j], false, false);
        }
    }

    // Bias + mask out-of-range columns with a large-negative finite sentinel.
    float bn[4];
    bool  ok[4];
    #pragma unroll
    for (int j = 0; j < 4; ++j) {
        const int n = colBase + j * 16 + l15;
        ok[j] = (n < N);
        bn[j] = ok[j] ? bias[n] : 0.0f;
    }

    // Per-row reduction. D layout: lane holds col=(l&15); VGPR r holds row
    // m = r + 8*half. xor-shuffles with mask<16 keep the halves independent.
    __shared__ float sM[8][32];
    __shared__ float sS[8][32];
    #pragma unroll
    for (int mt = 0; mt < 2; ++mt) {
        #pragma unroll
        for (int j = 0; j < 4; ++j)
            #pragma unroll
            for (int r = 0; r < 8; ++r) {
                float v = acc[mt][j][r] + bn[j];
                acc[mt][j][r] = ok[j] ? v : NEG_BIG;
            }
        #pragma unroll
        for (int r = 0; r < 8; ++r) {
            float m = fmaxf(fmaxf(acc[mt][0][r], acc[mt][1][r]),
                            fmaxf(acc[mt][2][r], acc[mt][3][r]));
            #pragma unroll
            for (int off = 1; off < 16; off <<= 1) m = fmaxf(m, __shfl_xor(m, off, 32));
            float s = __expf(acc[mt][0][r] - m) + __expf(acc[mt][1][r] - m)
                    + __expf(acc[mt][2][r] - m) + __expf(acc[mt][3][r] - m);
            #pragma unroll
            for (int off = 1; off < 16; off <<= 1) s += __shfl_xor(s, off, 32);
            if (l15 == 0) {
                const int row = mt * 16 + r + 8 * half;
                sM[wave][row] = m;
                sS[wave][row] = s;
            }
        }
    }
    __syncthreads();

    if (threadIdx.x < 32) {
        const int row = threadIdx.x;
        float M = NEG_BIG;
        #pragma unroll
        for (int w = 0; w < 8; ++w) M = fmaxf(M, sM[w][row]);
        float S = 0.0f;
        #pragma unroll
        for (int w = 0; w < 8; ++w) S += sS[w][row] * __expf(sM[w][row] - M);
        const size_t o = ((size_t)chunk * B + rowBase + row) * 2;
        partial[o]     = M;
        partial[o + 1] = S;
    }
}

// -----------------------------------------------------------------------------
// Fallback (small workspace): on-the-fly f32->bf16 conversion inside the GEMM.
// 16x64 tile per wave, v_wmma_f32_16x16x32_bf16.
// -----------------------------------------------------------------------------
__device__ __forceinline__ void load_cvt8(const float* __restrict__ p, v16bf& d, int o) {
    f32x4 a = *reinterpret_cast<const f32x4*>(p);
    f32x4 b = *reinterpret_cast<const f32x4*>(p + 4);
    d[o + 0] = (__bf16)a[0]; d[o + 1] = (__bf16)a[1];
    d[o + 2] = (__bf16)a[2]; d[o + 3] = (__bf16)a[3];
    d[o + 4] = (__bf16)b[0]; d[o + 5] = (__bf16)b[1];
    d[o + 6] = (__bf16)b[2]; d[o + 7] = (__bf16)b[3];
}

__global__ __launch_bounds__(256)
void gemm_lse_partial_f32(const float* __restrict__ X,
                          const float* __restrict__ W,
                          const float* __restrict__ bias,
                          int N, int K, int B,
                          float* __restrict__ partial)
{
    const int mTile  = blockIdx.x;
    const int chunk  = blockIdx.y;
    const int wave   = threadIdx.x >> 5;
    const int lane   = threadIdx.x & 31;
    const int half   = lane >> 4;
    const int l15    = lane & 15;

    const int rowBase = mTile * 16;
    const int colBase = chunk * 512 + wave * 64;

    v8f acc[4];
    #pragma unroll
    for (int j = 0; j < 4; ++j)
        #pragma unroll
        for (int r = 0; r < 8; ++r) acc[j][r] = 0.0f;

    const float* xrow = X + (size_t)(rowBase + l15) * K;
    const float* wrow[4];
    #pragma unroll
    for (int j = 0; j < 4; ++j) {
        int n = colBase + j * 16 + l15;
        if (n >= N) n = N - 1;
        wrow[j] = W + (size_t)n * K;
    }

    for (int ks = 0; ks < K / 32; ++ks) {
        const int kA = ks * 32 + 16 * half;
        v16bf afrag;
        load_cvt8(xrow + kA,      afrag, 0);
        load_cvt8(xrow + kA + 8,  afrag, 8);
        const int kB = ks * 32 + 16 * half;
        #pragma unroll
        for (int j = 0; j < 4; ++j) {
            __builtin_prefetch(wrow[j] + kB + 64, 0, 1);
            v16bf bfrag;
            load_cvt8(wrow[j] + kB,     bfrag, 0);
            load_cvt8(wrow[j] + kB + 8, bfrag, 8);
            acc[j] = __builtin_amdgcn_wmma_f32_16x16x32_bf16(
                false, afrag, false, bfrag, (short)0, acc[j], false, false);
        }
    }

    #pragma unroll
    for (int j = 0; j < 4; ++j) {
        const int n = colBase + j * 16 + l15;
        const float bv = (n < N) ? bias[n] : 0.0f;
        #pragma unroll
        for (int r = 0; r < 8; ++r) {
            float v = acc[j][r] + bv;
            acc[j][r] = (n < N) ? v : NEG_BIG;
        }
    }

    __shared__ float sM[8][16];
    __shared__ float sS[8][16];
    #pragma unroll
    for (int r = 0; r < 8; ++r) {
        float m = fmaxf(fmaxf(acc[0][r], acc[1][r]), fmaxf(acc[2][r], acc[3][r]));
        #pragma unroll
        for (int off = 1; off < 16; off <<= 1) m = fmaxf(m, __shfl_xor(m, off, 32));
        float s = __expf(acc[0][r] - m) + __expf(acc[1][r] - m)
                + __expf(acc[2][r] - m) + __expf(acc[3][r] - m);
        #pragma unroll
        for (int off = 1; off < 16; off <<= 1) s += __shfl_xor(s, off, 32);
        if (l15 == 0) {
            sM[wave][r + 8 * half] = m;
            sS[wave][r + 8 * half] = s;
        }
    }
    __syncthreads();

    if (threadIdx.x < 16) {
        const int row = threadIdx.x;
        float M = NEG_BIG;
        #pragma unroll
        for (int w = 0; w < 8; ++w) M = fmaxf(M, sM[w][row]);
        float S = 0.0f;
        #pragma unroll
        for (int w = 0; w < 8; ++w) S += sS[w][row] * __expf(sM[w][row] - M);
        const size_t o = ((size_t)chunk * B + rowBase + row) * 2;
        partial[o]     = M;
        partial[o + 1] = S;
    }
}

// -----------------------------------------------------------------------------
// Exact f32 dot products for the gathered target logits. One wave per row.
// -----------------------------------------------------------------------------
__global__ __launch_bounds__(256)
void gather_targets(const float* __restrict__ X, const int* __restrict__ T,
                    const float* __restrict__ w1,  const float* __restrict__ b1,
                    const float* __restrict__ w20, const float* __restrict__ b20,
                    const float* __restrict__ w21, const float* __restrict__ b21,
                    float* __restrict__ tgt, int B, int K)
{
    const int wave = threadIdx.x >> 5;
    const int lane = threadIdx.x & 31;
    const int row  = blockIdx.x * 8 + wave;
    if (row >= B) return;

    const int t = T[row];
    const int cluster = (t < 10000) ? -1 : ((t < 30000) ? 0 : 1);
    const float* x = X + (size_t)row * K;

    const int col1 = (cluster < 0) ? t : (10000 + cluster);
    const float* w = w1 + (size_t)col1 * K;
    float d1 = 0.0f;
    for (int e = lane; e < K; e += 32) d1 += x[e] * w[e];
    #pragma unroll
    for (int off = 16; off > 0; off >>= 1) d1 += __shfl_xor(d1, off, 32);

    float d2 = 0.0f;
    if (cluster >= 0) {
        const int off2  = 10000 + cluster * 20000;
        const float* w2 = (cluster ? w21 : w20) + (size_t)(t - off2) * K;
        for (int e = lane; e < K; e += 32) d2 += x[e] * w2[e];
        #pragma unroll
        for (int off = 16; off > 0; off >>= 1) d2 += __shfl_xor(d2, off, 32);
        if (lane == 0) d2 += (cluster ? b21 : b20)[t - off2];
    }
    if (lane == 0) {
        tgt[(size_t)row * 2 + 0] = d1 + b1[col1];
        tgt[(size_t)row * 2 + 1] = d2;
    }
}

// -----------------------------------------------------------------------------
// Merge chunk partials -> lse, per-row loss, block reduce, atomicAdd(out, /B).
// -----------------------------------------------------------------------------
__device__ __forceinline__ float merge_lse(const float* __restrict__ p,
                                           int nc, int B, int row)
{
    float M = NEG_BIG, S = 0.0f;
    for (int c = 0; c < nc; ++c) {
        const size_t o = ((size_t)c * B + row) * 2;
        const float m = p[o], s = p[o + 1];
        if (m > M) { S = S * __expf(M - m) + s; M = m; }
        else       { S += s * __expf(m - M); }
    }
    return M + __logf(S);
}

__global__ __launch_bounds__(256)
void finalize_loss(const int* __restrict__ T,
                   const float* __restrict__ p1,
                   const float* __restrict__ p20,
                   const float* __restrict__ p21,
                   const float* __restrict__ tgt,
                   float* __restrict__ out,
                   int B, int NC1, int NC2)
{
    const int row = blockIdx.x * blockDim.x + threadIdx.x;
    float loss = 0.0f;
    if (row < B) {
        const float lse1 = merge_lse(p1, NC1, B, row);
        loss = -(tgt[(size_t)row * 2 + 0] - lse1);
        const int t = T[row];
        const int cluster = (t < 10000) ? -1 : ((t < 30000) ? 0 : 1);
        if (cluster >= 0) {
            const float* p2 = cluster ? p21 : p20;
            const float lse2 = merge_lse(p2, NC2, B, row);
            loss += -(tgt[(size_t)row * 2 + 1] - lse2);
        }
    }
    __shared__ float red[256];
    red[threadIdx.x] = loss;
    __syncthreads();
    for (int s = 128; s > 0; s >>= 1) {
        if ((int)threadIdx.x < s) red[threadIdx.x] += red[threadIdx.x + s];
        __syncthreads();
    }
    if (threadIdx.x == 0) atomicAdd(out, red[0] / (float)B);
}

// -----------------------------------------------------------------------------
extern "C" void kernel_launch(void* const* d_in, const int* in_sizes, int n_in,
                              void* d_out, int out_size, void* d_ws, size_t ws_size,
                              hipStream_t stream) {
    const float* x   = (const float*)d_in[0];
    const int*   t   = (const int*)  d_in[1];
    const float* w1  = (const float*)d_in[2];
    const float* b1  = (const float*)d_in[3];
    const float* w20 = (const float*)d_in[4];
    const float* b20 = (const float*)d_in[5];
    const float* w21 = (const float*)d_in[6];
    const float* b21 = (const float*)d_in[7];
    float* out = (float*)d_out;

    const int B  = in_sizes[1];          // 4096
    const int K  = in_sizes[0] / B;      // 512
    const int N1 = in_sizes[3];          // 10002
    const int N2 = in_sizes[5];          // 20000
    const int NC1 = (N1 + 511) / 512;    // 20
    const int NC2 = (N2 + 511) / 512;    // 40

    float* p1  = (float*)d_ws;                       // [NC1][B][2]
    float* p20 = p1  + (size_t)NC1 * B * 2;          // [NC2][B][2]
    float* p21 = p20 + (size_t)NC2 * B * 2;          // [NC2][B][2]
    float* tgt = p21 + (size_t)NC2 * B * 2;          // [B][2]
    char*  heap = (char*)(tgt + (size_t)B * 2);

    const size_t base_bytes = (size_t)(heap - (char*)d_ws);
    const size_t conv_elems = (size_t)B * K + (size_t)N1 * K + 2 * (size_t)N2 * K;
    const bool use_f16 = (ws_size >= base_bytes + conv_elems * sizeof(_Float16) + 256);

    hipMemsetAsync(d_out, 0, (size_t)out_size * sizeof(float), stream);

    dim3 blk(256);
    if (use_f16) {
        _Float16* xh   = (_Float16*)(((uintptr_t)heap + 255) & ~(uintptr_t)255);
        _Float16* wh1  = xh   + (size_t)B  * K;
        _Float16* wh20 = wh1  + (size_t)N1 * K;
        _Float16* wh21 = wh20 + (size_t)N2 * K;

        auto cvt = [&](const float* s, _Float16* d, size_t n) {
            int g = (int)((n / 8 + 255) / 256);
            cvt_f32_f16<<<dim3(g), blk, 0, stream>>>(s, d, (long long)n);
        };
        cvt(x,   xh,   (size_t)B  * K);
        cvt(w1,  wh1,  (size_t)N1 * K);
        cvt(w20, wh20, (size_t)N2 * K);
        cvt(w21, wh21, (size_t)N2 * K);

        gemm_lse_partial_h<<<dim3(B / 32, NC1), blk, 0, stream>>>(xh, wh1,  b1,  N1, K, B, p1);
        gemm_lse_partial_h<<<dim3(B / 32, NC2), blk, 0, stream>>>(xh, wh20, b20, N2, K, B, p20);
        gemm_lse_partial_h<<<dim3(B / 32, NC2), blk, 0, stream>>>(xh, wh21, b21, N2, K, B, p21);
    } else {
        gemm_lse_partial_f32<<<dim3(B / 16, NC1), blk, 0, stream>>>(x, w1,  b1,  N1, K, B, p1);
        gemm_lse_partial_f32<<<dim3(B / 16, NC2), blk, 0, stream>>>(x, w20, b20, N2, K, B, p20);
        gemm_lse_partial_f32<<<dim3(B / 16, NC2), blk, 0, stream>>>(x, w21, b21, N2, K, B, p21);
    }
    gather_targets<<<dim3((B + 7) / 8), blk, 0, stream>>>(x, t, w1, b1, w20, b20, w21, b21, tgt, B, K);
    finalize_loss<<<dim3((B + 255) / 256), blk, 0, stream>>>(t, p1, p20, p21, tgt, out, B, NC1, NC2);
}
